// Attention_21500606284159
// MI455X (gfx1250) — compile-verified, matrix operand order checked by reading, not run on hardware
//
#include <hip/hip_runtime.h>

typedef _Float16 v4h  __attribute__((ext_vector_type(4)));
typedef _Float16 v8h  __attribute__((ext_vector_type(8)));
typedef _Float16 v16h __attribute__((ext_vector_type(16)));
typedef float    v8f  __attribute__((ext_vector_type(8)));

#define B_     64
#define T_     2048
#define E_     1024
#define D_     1024
#define FANIN_ 2048
#define LDA_   1032   // padded f16 stride: 2064B = 129*16B -> 16B-aligned rows, bank-spread
#define TROWS_ 64     // t-rows per block (A-reuse factor for each B read)
#define CH_    16     // t-rows per async chunk in k_context

// ---------------- K0: convert We = W[:, :1024] to f16 ----------------
__global__ void k_convert_we(const float* __restrict__ W, _Float16* __restrict__ We_h) {
    int i = blockIdx.x * blockDim.x + threadIdx.x;   // 0 .. 1024*1024-1
    int d = i >> 10, e = i & 1023;
    We_h[i] = (_Float16)W[(size_t)d * FANIN_ + e];
}

// ---------------- K1: s[b,d] = Wd[d,:] . dec[b,:] + bias[d] ----------------
__global__ void k_dec_proj(const float* __restrict__ W, const float* __restrict__ bias,
                           const float* __restrict__ dec, float* __restrict__ s) {
    __shared__ float decs[D_];
    int b = blockIdx.x;
    for (int i = threadIdx.x; i < D_; i += blockDim.x) decs[i] = dec[b * D_ + i];
    __syncthreads();
    int d = blockIdx.y * blockDim.x + threadIdx.x;
    const float* wrow = W + (size_t)d * FANIN_ + E_;
    float acc = bias[d];
    #pragma unroll 8
    for (int e = 0; e < D_; ++e) acc = fmaf(wrow[e], decs[e], acc);
    s[b * D_ + d] = acc;
}

// ---------------- K2: fused projection-GEMM + tanh + v-dot -> scores ----------------
// Block: 512 threads = 16 waves, covers 64 t-rows x 1024 d.
// Wave w: d-range [w*64, w*64+64) (4 N-tiles), all 4 t-tiles.
__global__ __launch_bounds__(512) void k_scores(const float* __restrict__ enc,
                                                const _Float16* __restrict__ We_h,
                                                const float* __restrict__ s,
                                                const float* __restrict__ v,
                                                float* __restrict__ scores) {
    __shared__ _Float16 lds_a[TROWS_ * LDA_];   // 64 x 1024 K, f16, padded: 132 KB
    __shared__ float    sc[TROWS_];

    const int tid = threadIdx.x;
    const int b   = blockIdx.x >> 5;          // 32 t-blocks per batch
    const int t0  = (blockIdx.x & 31) << 6;   // 64 rows of T per block

    if (tid < TROWS_) sc[tid] = 0.0f;

    // Stage A tile: 64x1024 f32 -> f16 in LDS (coalesced global, b64 LDS stores)
    const float* abase = enc + ((size_t)(b * T_ + t0)) * E_;
    for (int i = tid; i < TROWS_ * 1024 / 4; i += 512) {
        float4 f = ((const float4*)abase)[i];
        int idx = i * 4;
        int r = idx >> 10, c = idx & 1023;
        v4h hv = { (_Float16)f.x, (_Float16)f.y, (_Float16)f.z, (_Float16)f.w };
        *(v4h*)&lds_a[r * LDA_ + c] = hv;
    }
    __syncthreads();

    const int w    = tid >> 5;        // wave id 0..15: owns 64 d-columns
    const int lane = tid & 31;
    const int half = lane >> 4;       // ISA 16-bit A/B layout: lanes 16-31 hold K+8 chunks
    const int l16  = lane & 15;
    const int koff = half * 8;
    const int dbase = w * 64;

    v8f zero = {};
    v8f acc[4][4];                    // [t-tile][N-tile] : 128 VGPRs
    #pragma unroll
    for (int tt = 0; tt < 4; ++tt)
        #pragma unroll
        for (int j = 0; j < 4; ++j) acc[tt][j] = zero;

    for (int k0 = 0; k0 < E_; k0 += 32) {
        // A fragments from LDS: lane holds row l16 of its tile,
        // K chunks [k0+koff, +8) and [k0+16+koff, +8)
        v16h a[4];
        #pragma unroll
        for (int tt = 0; tt < 4; ++tt) {
            const _Float16* ap = &lds_a[(tt * 16 + l16) * LDA_ + k0 + koff];
            union { v16h v; v8h h[2]; } u;
            u.h[0] = *(const v8h*)ap;
            u.h[1] = *(const v8h*)(ap + 16);
            a[tt] = u.v;
        }
        // B fragments: column N=l16 <-> row d of We_h (contiguous along K, L2-hot)
        #pragma unroll
        for (int j = 0; j < 4; ++j) {
            const _Float16* bp = We_h + (size_t)(dbase + j * 16 + l16) * E_ + k0 + koff;
            union { v16h v; v8h h[2]; } u;
            u.h[0] = *(const v8h*)bp;
            u.h[1] = *(const v8h*)(bp + 16);
            v16h bb = u.v;
            #pragma unroll
            for (int tt = 0; tt < 4; ++tt)
                acc[tt][j] = __builtin_amdgcn_wmma_f32_16x16x32_f16(
                    false, a[tt], false, bb, (short)0, acc[tt][j], false, false);
        }
    }

    // Epilogue: score_part[t] += v[d] * tanh(P[t,d] + s[b,d])
    // C layout: VGPR r, lane -> M = r + 8*half, N = l16
    float part[4][8];
    #pragma unroll
    for (int tt = 0; tt < 4; ++tt)
        #pragma unroll
        for (int r = 0; r < 8; ++r) part[tt][r] = 0.0f;

    #pragma unroll
    for (int j = 0; j < 4; ++j) {
        int d = dbase + j * 16 + l16;
        float sv = s[b * D_ + d];
        float vv = v[d];
        #pragma unroll
        for (int tt = 0; tt < 4; ++tt)
            #pragma unroll
            for (int r = 0; r < 8; ++r)
                part[tt][r] += vv * tanhf(acc[tt][j][r] + sv);
    }

    // Reduce across the 16 N-lanes of each half, then across waves via LDS atomics
    #pragma unroll
    for (int tt = 0; tt < 4; ++tt)
        #pragma unroll
        for (int r = 0; r < 8; ++r) {
            float x = part[tt][r];
            x += __shfl_xor(x, 1);
            x += __shfl_xor(x, 2);
            x += __shfl_xor(x, 4);
            x += __shfl_xor(x, 8);
            if (l16 == 0) atomicAdd(&sc[tt * 16 + half * 8 + r], x);
        }
    __syncthreads();
    if (tid < TROWS_) scores[(size_t)b * T_ + t0 + tid] = sc[tid];
}

// ---------------- K3: softmax over T per batch ----------------
__global__ void k_softmax(const float* __restrict__ scores, float* __restrict__ alpha) {
    __shared__ float red[256];
    int b = blockIdx.x;
    const float* srow = scores + (size_t)b * T_;
    float vals[8];
    float m = -1e30f;
    #pragma unroll
    for (int i = 0; i < 8; ++i) { vals[i] = srow[i * 256 + threadIdx.x]; m = fmaxf(m, vals[i]); }
    red[threadIdx.x] = m; __syncthreads();
    for (int off = 128; off > 0; off >>= 1) {
        if (threadIdx.x < off) red[threadIdx.x] = fmaxf(red[threadIdx.x], red[threadIdx.x + off]);
        __syncthreads();
    }
    m = red[0]; __syncthreads();
    float sum = 0.0f;
    #pragma unroll
    for (int i = 0; i < 8; ++i) { vals[i] = __expf(vals[i] - m); sum += vals[i]; }
    red[threadIdx.x] = sum; __syncthreads();
    for (int off = 128; off > 0; off >>= 1) {
        if (threadIdx.x < off) red[threadIdx.x] += red[threadIdx.x + off];
        __syncthreads();
    }
    float inv = 1.0f / red[0];
    #pragma unroll
    for (int i = 0; i < 8; ++i)
        alpha[(size_t)b * T_ + i * 256 + threadIdx.x] = vals[i] * inv;
}

// ---------------- K4: context = sum_t alpha[b,t] * enc[b,t,:] ----------------
__global__ void k_zero(float* __restrict__ out, int n) {
    int i = blockIdx.x * blockDim.x + threadIdx.x;
    if (i < n) out[i] = 0.0f;
}

// Double-buffered async global->LDS pipeline: chunk c+1 streams into LDS on the
// ASYNCcnt path while chunk c is reduced out of LDS.
__global__ __launch_bounds__(256) void k_context(const float* __restrict__ enc,
                                                 const float* __restrict__ alpha,
                                                 float* __restrict__ out) {
    __shared__ __align__(16) float buf[2][CH_ * 256];   // 2 x 16 rows x 256 cols = 32 KB
    __shared__ float al[256];

    const int tid = threadIdx.x;
    const int b   = blockIdx.x;
    const int e0  = blockIdx.y * 256;
    const int t0  = blockIdx.z * 256;

    al[tid] = alpha[(size_t)b * T_ + t0 + tid];

    const float* gbase = enc + ((size_t)(b * T_ + t0)) * E_ + e0;

    // Issue one chunk: 16 rows x 256 f32 = 1024 b128 transfers, 4 per thread.
    auto fill = [&](int c) {
        const int bi = c & 1;
        #pragma unroll
        for (int i = 0; i < 4; ++i) {
            int q    = i * 256 + tid;   // 0..1023
            int row  = q >> 6;          // 0..15
            int col4 = q & 63;          // b128 slot within row
            unsigned lds_off = (unsigned)(size_t)&buf[bi][row * 256 + col4 * 4];
            unsigned long long ga =
                (unsigned long long)(size_t)(gbase + (size_t)(c * CH_ + row) * E_ + col4 * 4);
            asm volatile("global_load_async_to_lds_b128 %0, %1, off"
                         :: "v"(lds_off), "v"(ga) : "memory");
        }
    };

    float acc = 0.0f;
    fill(0);
    for (int c = 0; c < 256 / CH_; ++c) {
        if (c + 1 < 256 / CH_) {
            fill(c + 1);                                    // 4 more in flight
            asm volatile("s_wait_asynccnt 0x4" ::: "memory"); // chunk c fully landed
        } else {
            asm volatile("s_wait_asynccnt 0x0" ::: "memory");
        }
        __syncthreads();                                    // visible to all waves
        const int bi = c & 1;
        #pragma unroll
        for (int r = 0; r < CH_; ++r)
            acc = fmaf(al[c * CH_ + r], buf[bi][r * 256 + tid], acc);
        __syncthreads();                                    // done reading before overwrite
    }
    atomicAdd(&out[b * E_ + e0 + tid], acc);
}

// ---------------- launcher ----------------
extern "C" void kernel_launch(void* const* d_in, const int* in_sizes, int n_in,
                              void* d_out, int out_size, void* d_ws, size_t ws_size,
                              hipStream_t stream) {
    const float* enc  = (const float*)d_in[0];   // (64, 2048, 1024)
    const float* dec  = (const float*)d_in[1];   // (64, 1024)
    const float* W    = (const float*)d_in[2];   // (1024, 2048)
    const float* bias = (const float*)d_in[3];   // (1024,)
    const float* v    = (const float*)d_in[4];   // (1, 1024)
    float* out = (float*)d_out;                  // (64, 1024)

    char* ws = (char*)d_ws;
    _Float16* We_h  = (_Float16*)ws;                                   // 2 MB
    float*    s     = (float*)(ws + (2u << 20));                       // 256 KB
    float*    score = (float*)(ws + (2u << 20) + (256u << 10));        // 512 KB
    float*    alpha = (float*)(ws + (2u << 20) + (768u << 10));        // 512 KB

    k_convert_we<<<4096, 256, 0, stream>>>(W, We_h);
    k_dec_proj<<<dim3(64, 4), 256, 0, stream>>>(W, bias, dec, s);
    k_scores<<<B_ * (T_ / TROWS_), 512, 0, stream>>>(enc, We_h, s, v, score);
    k_softmax<<<64, 256, 0, stream>>>(score, alpha);
    k_zero<<<(B_ * D_ + 255) / 256, 256, 0, stream>>>(out, B_ * D_);
    k_context<<<dim3(64, 4, 8), 256, 0, stream>>>(enc, alpha, out);
}
